// xLSTMModel_45698452029459
// MI455X (gfx1250) — matrix operand order repaired and checked by reading, not compiled, not run on hardware
//
#include <hip/hip_runtime.h>
#include <math.h>

// ---------------- model dims ----------------
#define EMB    640
#define NFEATS 64
#define SWIN   128
#define NB32   32
#define DINNER 1280
#define NHM    4
#define DHM    320
#define NBQ    256
#define BSQ    5
#define NHS    4
#define DHS    160
#define PFF    832

typedef __attribute__((ext_vector_type(16))) _Float16 v16h;
typedef __attribute__((ext_vector_type(8)))  float    v8f;

// ---------------- device math helpers ----------------
__device__ __forceinline__ float siluf(float x){ return x / (1.f + expf(-x)); }
__device__ __forceinline__ float sigmf(float x){ return 1.f / (1.f + expf(-x)); }
__device__ __forceinline__ float logsigf(float x){ return fminf(x, 0.f) - log1pf(expf(-fabsf(x))); }
__device__ __forceinline__ float geluf(float x){ return 0.5f * x * (1.f + erff(x * 0.7071067811865475f)); }

#define GSTRIDE(i, total) \
  for (long i = (long)blockIdx.x * blockDim.x + threadIdx.x; i < (total); \
       i += (long)gridDim.x * blockDim.x)

// ---------------- WMMA fragment loaders (16-bit, 16x16x32, wave32) ----------------
// A (16x32): lanes 0-15 -> M=lane, K in {0..7,16..23}; lanes 16-31 -> M=lane-16, K in {8..15,24..31}
__device__ __forceinline__ v16h frag_a(const _Float16* base, int ld, int mbase, int lane)
{
  const int m  = mbase + (lane & 15);
  const int kh = (lane & 16) ? 8 : 0;
  const _Float16* p = base + (long)m * ld + kh;
  v16h a;
  #pragma unroll
  for (int j = 0; j < 4; ++j) { a[2*j]   = p[2*j];    a[2*j+1]   = p[2*j+1]; }
  #pragma unroll
  for (int j = 0; j < 4; ++j) { a[8+2*j] = p[16+2*j]; a[8+2*j+1] = p[16+2*j+1]; }
  return a;
}
// B (32x16) stored LDS-side as [n][k]: lanes 0-15 -> N=lane, K in {0..7,16..23}; lanes 16-31 -> N=lane-16, K in {8..15,24..31}
__device__ __forceinline__ v16h frag_b(const _Float16* base, int ld, int nbase, int lane)
{
  const int n  = nbase + (lane & 15);
  const int kb = (lane & 16) ? 8 : 0;
  const _Float16* p = base + (long)n * ld + kb;
  v16h b;
  #pragma unroll
  for (int j = 0; j < 8; ++j) { b[j] = p[j]; b[8+j] = p[16+j]; }
  return b;
}

// ---------------- generic batched WMMA GEMM ----------------
// C[M,N] = alpha * A[M,K] x op(B) + bias ;  bKN==0: B is [N,K] (A*B^T), bKN==1: B is [K,N] (A*B)
#define GT_M 64
#define GT_N 64
#define GT_K 32
#define GLDS 40

__global__ __launch_bounds__(128)
void k_gemm(const float* __restrict__ A, long strideA, int lda,
            const float* __restrict__ B, long strideB, int ldb, int bKN,
            float* __restrict__ C, long strideC, int ldc,
            const float* __restrict__ bias, long strideBias,
            int M, int N, int K, float alpha)
{
  __shared__ _Float16 sA[GT_M * GLDS];
  __shared__ _Float16 sB[GT_N * GLDS];
  const int bz = blockIdx.z;
  const float* Ab = A + (long)bz * strideA;
  const float* Bb = B + (long)bz * strideB;
  float* Cb = C + (long)bz * strideC;
  const float* biasb = bias ? (bias + (long)bz * strideBias) : nullptr;
  const int tileM = blockIdx.y * GT_M;
  const int tileN = blockIdx.x * GT_N;
  const int tid  = threadIdx.x;
  const int wave = tid >> 5;
  const int lane = tid & 31;
  const int wm = (wave >> 1) * 32;
  const int wn = (wave & 1) * 32;

  v8f acc[2][2];
  #pragma unroll
  for (int i = 0; i < 2; ++i)
    #pragma unroll
    for (int j = 0; j < 2; ++j)
      #pragma unroll
      for (int r = 0; r < 8; ++r) acc[i][j][r] = 0.f;

  for (int k0 = 0; k0 < K; k0 += GT_K) {
    #pragma unroll
    for (int it = 0; it < 16; ++it) {
      int idx = it * 128 + tid;
      int row = idx >> 5, col = idx & 31;
      int gm = tileM + row, gk = k0 + col;
      float va = (gm < M && gk < K) ? Ab[(long)gm * lda + gk] : 0.f;
      sA[row * GLDS + col] = (_Float16)va;
      int gn = tileN + row;
      float vb;
      if (bKN == 0) vb = (gn < N && gk < K) ? Bb[(long)gn * ldb + gk] : 0.f;
      else          vb = (gn < N && gk < K) ? Bb[(long)gk * ldb + gn] : 0.f;
      sB[row * GLDS + col] = (_Float16)vb;
    }
    __syncthreads();
    if (k0 + GT_K < K && tid < 64) {       // prefetch next K-slab (global_prefetch_b8)
      int gm = tileM + tid;
      if (gm < M) __builtin_prefetch(&Ab[(long)gm * lda + k0 + GT_K], 0, 0);
    }
    v16h bf0 = frag_b(sB, GLDS, wn,      lane);
    v16h bf1 = frag_b(sB, GLDS, wn + 16, lane);
    #pragma unroll
    for (int mt = 0; mt < 2; ++mt) {
      v16h af = frag_a(sA, GLDS, wm + mt * 16, lane);
      acc[mt][0] = __builtin_amdgcn_wmma_f32_16x16x32_f16(false, af, false, bf0, (short)0, acc[mt][0], false, false);
      acc[mt][1] = __builtin_amdgcn_wmma_f32_16x16x32_f16(false, af, false, bf1, (short)0, acc[mt][1], false, false);
    }
    __syncthreads();
  }

  const int lmod = lane & 15;
  const int lhi8 = (lane & 16) ? 8 : 0;
  #pragma unroll
  for (int mt = 0; mt < 2; ++mt)
    #pragma unroll
    for (int nt = 0; nt < 2; ++nt) {
      int n = tileN + wn + nt * 16 + lmod;
      if (n >= N) continue;
      float bv = biasb ? biasb[n] : 0.f;
      int mb = tileM + wm + mt * 16 + lhi8;
      #pragma unroll
      for (int r = 0; r < 8; ++r) {
        int m = mb + r;
        if (m < M) Cb[(long)m * ldc + n] = acc[mt][nt][r] * alpha + bv;
      }
    }
}

// ---------------- sLSTM recurrent scan (one WG per head, R resident in 320KB LDS) ----------------
// gx: (4 gates, NHS, 32*S, 160) ; recw: (4, NHS, 160, 160) ; ys out: (32, S, NHS*160)
__global__ __launch_bounds__(320)
void k_slstm_scan(const float* __restrict__ gx,
                  const float* __restrict__ recw,
                  float* __restrict__ ys,
                  int S)
{
  extern __shared__ char smem[];
  _Float16* Rl = (_Float16*)smem;           // 4*160*160 f16 = 200KB
  _Float16* yl = Rl + 4 * DHS * DHS;        // 32*160   f16 = 10KB
  const int head = blockIdx.x;
  const int tid  = threadIdx.x;
  const int wave = tid >> 5, lane = tid & 31;
  const long rows = (long)NB32 * S;

  for (int i = tid; i < 4 * DHS * DHS; i += blockDim.x) {
    int g = i / (DHS * DHS); int rem = i - g * DHS * DHS;
    Rl[i] = (_Float16)recw[((long)(g * NHS + head)) * DHS * DHS + rem];
  }
  for (int i = tid; i < NB32 * DHS; i += blockDim.x) yl[i] = (_Float16)0.f;
  __syncthreads();

  float stc[2][8], stn[2][8], stm[2][8];
  #pragma unroll
  for (int p = 0; p < 2; ++p)
    #pragma unroll
    for (int r = 0; r < 8; ++r) { stc[p][r] = 0.f; stn[p][r] = 0.f; stm[p][r] = 0.f; }

  const int lmod = lane & 15;
  const int lhi8 = (lane & 16) ? 8 : 0;

  for (int t = 0; t < S; ++t) {
    v8f acc[2][4];
    #pragma unroll
    for (int pi = 0; pi < 2; ++pi) {
      const int pos = wave * 2 + pi;       // 0..19 -> 2 mtiles x 10 ntiles
      const int mt = pos / 10, nt = pos - (pos / 10) * 10;
      const int d = nt * 16 + lmod;
      #pragma unroll
      for (int g = 0; g < 4; ++g) {
        const float* gp = gx + ((long)(g * NHS + head)) * rows * DHS;
        #pragma unroll
        for (int r = 0; r < 8; ++r) {
          int b = mt * 16 + lhi8 + r;
          acc[pi][g][r] = gp[((long)b * S + t) * DHS + d];
        }
      }
    }
    #pragma unroll
    for (int pi = 0; pi < 2; ++pi) {
      const int pos = wave * 2 + pi;
      const int mt = pos / 10, nt = pos - (pos / 10) * 10;
      for (int kk = 0; kk < 5; ++kk) {
        v16h a = frag_a(yl + kk * 32, DHS, mt * 16, lane);
        #pragma unroll
        for (int g = 0; g < 4; ++g) {
          v16h b = frag_b(Rl + g * DHS * DHS + kk * 32, DHS, nt * 16, lane);
          acc[pi][g] = __builtin_amdgcn_wmma_f32_16x16x32_f16(false, a, false, b, (short)0, acc[pi][g], false, false);
        }
      }
    }
    __syncthreads();                        // all reads of y(t-1) done
    #pragma unroll
    for (int pi = 0; pi < 2; ++pi) {
      const int pos = wave * 2 + pi;
      const int mt = pos / 10, nt = pos - (pos / 10) * 10;
      const int d = nt * 16 + lmod;
      #pragma unroll
      for (int r = 0; r < 8; ++r) {
        int b = mt * 16 + lhi8 + r;
        float iraw = acc[pi][0][r], fraw = acc[pi][1][r];
        float zraw = acc[pi][2][r], oraw = acc[pi][3][r];
        float lf   = logsigf(fraw) + stm[pi][r];
        float mnew = fmaxf(iraw, lf);
        float ii   = expf(iraw - mnew);
        float ff   = expf(lf - mnew);
        float cnew = ff * stc[pi][r] + ii * tanhf(zraw);
        float nnew = ff * stn[pi][r] + ii;
        stc[pi][r] = cnew; stn[pi][r] = nnew; stm[pi][r] = mnew;
        float yv = sigmf(oraw) * cnew / nnew;
        yl[b * DHS + d] = (_Float16)yv;
        ys[(((long)b * S + t) * NHS + head) * DHS + d] = yv;
      }
    }
    __syncthreads();
  }
}

// ---------------- elementwise / reduction kernels ----------------
__global__ void k_layernorm(const float* __restrict__ x, const float* __restrict__ w,
                            float* __restrict__ y, int n)
{
  const int row = blockIdx.x;
  const float* xr = x + (long)row * n;
  float s = 0.f, ss = 0.f;
  for (int i = threadIdx.x; i < n; i += blockDim.x) { float v = xr[i]; s += v; ss += v * v; }
  __shared__ float sh1[256], sh2[256];
  sh1[threadIdx.x] = s; sh2[threadIdx.x] = ss; __syncthreads();
  for (int o = blockDim.x >> 1; o > 0; o >>= 1) {
    if (threadIdx.x < o) { sh1[threadIdx.x] += sh1[threadIdx.x + o]; sh2[threadIdx.x] += sh2[threadIdx.x + o]; }
    __syncthreads();
  }
  float mean = sh1[0] / n;
  float inv  = rsqrtf(sh2[0] / n - mean * mean + 1e-5f);
  float* yr = y + (long)row * n;
  for (int i = threadIdx.x; i < n; i += blockDim.x) yr[i] = (xr[i] - mean) * inv * w[i];
}

// per-(b,s,head) norm over DH.  layout 0: h=(b*NH+hh, S, DH)   layout 1: h=(b*S+s, NH, DH)
__global__ void k_mhnorm(const float* __restrict__ h, const float* __restrict__ w,
                         float* __restrict__ y, int S, int NH, int DH, int layout)
{
  const int idx = blockIdx.x;
  const int hh  = idx % NH;
  const long bs = idx / NH;
  const int s = (int)(bs % S); const int b = (int)(bs / S);
  const float* src = (layout == 0)
    ? h + (((long)(b * NH + hh)) * S + s) * DH
    : h + ((long)bs * NH + hh) * DH;
  float sum = 0.f, ssum = 0.f;
  for (int i = threadIdx.x; i < DH; i += blockDim.x) { float v = src[i]; sum += v; ssum += v * v; }
  __shared__ float sh1[128], sh2[128];
  sh1[threadIdx.x] = sum; sh2[threadIdx.x] = ssum; __syncthreads();
  for (int o = blockDim.x >> 1; o > 0; o >>= 1) {
    if (threadIdx.x < o) { sh1[threadIdx.x] += sh1[threadIdx.x + o]; sh2[threadIdx.x] += sh2[threadIdx.x + o]; }
    __syncthreads();
  }
  float mean = sh1[0] / DH;
  float inv  = rsqrtf(sh2[0] / DH - mean * mean + 1e-5f);
  float* dst = y + (long)bs * NH * DH + (long)hh * DH;
  for (int i = threadIdx.x; i < DH; i += blockDim.x) dst[i] = (src[i] - mean) * inv * w[hh * DH + i];
}

// depthwise causal conv + SiLU; x,y: (B,S,C) channel-major rows
__global__ void k_conv_silu(const float* __restrict__ x, const float* __restrict__ w,
                            const float* __restrict__ bias, float* __restrict__ y,
                            int B, int S, int C, int K)
{
  GSTRIDE(i, (long)B * S * C) {
    int c = (int)(i % C); long bs = i / C; int s = (int)(bs % S); int b = (int)(bs / S);
    float acc = bias[c];
    const float* xb = x + ((long)b * S) * C + c;
    for (int kk = 0; kk < K; ++kk) {
      int t = s - (K - 1) + kk;
      if (t >= 0) acc += xb[(long)t * C] * w[c * K + kk];
    }
    y[i] = siluf(acc);
  }
}

// block-diagonal headwise matmul: out[r, nb*bo+o] = sum_i x[r, nb*bi+i] * w[nb,o,i]
__global__ void k_headwise(const float* __restrict__ x, const float* __restrict__ w,
                           float* __restrict__ y, long rows, int nb, int bo, int bi)
{
  GSTRIDE(i, rows * nb * bo) {
    int o = (int)(i % bo); long t = i / bo; int n = (int)(t % nb); long r = t / nb;
    const float* xr = x + r * (long)nb * bi + (long)n * bi;
    const float* wr = w + ((long)n * bo + o) * bi;
    float acc = 0.f;
    for (int j = 0; j < bi; ++j) acc += xr[j] * wr[j];
    y[i] = acc;
  }
}

// ig/fg gates: ifin = [q|k|v]; ig[(b*4+h)*S+s], fg likewise
__global__ void k_mlstm_gates(const float* __restrict__ q, const float* __restrict__ k,
                              const float* __restrict__ v,
                              const float* __restrict__ igw, const float* __restrict__ igb,
                              const float* __restrict__ fgw, const float* __restrict__ fgb,
                              float* __restrict__ ig, float* __restrict__ fg, int S, int C)
{
  const int row = blockIdx.x;
  const int s = row % S, b = row / S;
  const float* qr = q + (long)row * C;
  const float* kr = k + (long)row * C;
  const float* vr = v + (long)row * C;
  float aI[4] = {0,0,0,0}, aF[4] = {0,0,0,0};
  for (int i = threadIdx.x; i < C; i += blockDim.x) {
    float qv = qr[i], kv = kr[i], vv = vr[i];
    #pragma unroll
    for (int h = 0; h < 4; ++h) {
      const float* wi = igw + (long)h * 3 * C;
      aI[h] += qv * wi[i] + kv * wi[C + i] + vv * wi[2 * C + i];
      const float* wf = fgw + (long)h * 3 * C;
      aF[h] += qv * wf[i] + kv * wf[C + i] + vv * wf[2 * C + i];
    }
  }
  __shared__ float sh[256];
  for (int h = 0; h < 4; ++h) {
    sh[threadIdx.x] = aI[h]; __syncthreads();
    for (int o = blockDim.x >> 1; o > 0; o >>= 1) { if (threadIdx.x < o) sh[threadIdx.x] += sh[threadIdx.x + o]; __syncthreads(); }
    if (threadIdx.x == 0) ig[((long)b * 4 + h) * S + s] = sh[0] + igb[h];
    __syncthreads();
    sh[threadIdx.x] = aF[h]; __syncthreads();
    for (int o = blockDim.x >> 1; o > 0; o >>= 1) { if (threadIdx.x < o) sh[threadIdx.x] += sh[threadIdx.x + o]; __syncthreads(); }
    if (threadIdx.x == 0) fg[((long)b * 4 + h) * S + s] = sh[0] + fgb[h];
    __syncthreads();
  }
}

__global__ void k_cumlogsig(const float* __restrict__ fg, float* __restrict__ dcum,
                            int rowsTot, int S)
{
  int r = blockIdx.x * blockDim.x + threadIdx.x;
  if (r >= rowsTot) return;
  const float* f = fg + (long)r * S; float* o = dcum + (long)r * S;
  float c = 0.f;
  for (int s = 0; s < S; ++s) { c += logsigf(f[s]); o[s] = c; }
}

// (B,S,NH*DH) -> (B*NH, S, DH)
__global__ void k_pack_heads(const float* __restrict__ x, float* __restrict__ y,
                             int B, int S, int NH, int DH)
{
  GSTRIDE(i, (long)B * S * NH * DH) {
    int d = (int)(i % DH); long t = i / DH; int s = (int)(t % S); long bh = t / S;
    int h = (int)(bh % NH); int b = (int)(bh / NH);
    y[i] = x[((long)b * S + s) * (long)NH * DH + (long)h * DH + d];
  }
}

// in-place decay-matrix weighting + row normalizer on C (BH,S,S); C already scaled by DH^-0.5
__global__ void k_attn_norm(float* __restrict__ C, const float* __restrict__ dcum,
                            const float* __restrict__ ig, int S)
{
  const int i  = blockIdx.x % S;
  const int bh = blockIdx.x / S;
  float* row = C + ((long)bh * S + i) * S;
  const float* dc  = dcum + (long)bh * S;
  const float* igr = ig   + (long)bh * S;
  const float di = dc[i];
  __shared__ float sh[128];
  float mx = -3.0e38f;
  for (int j = threadIdx.x; j <= i; j += blockDim.x) mx = fmaxf(mx, di - dc[j] + igr[j]);
  sh[threadIdx.x] = mx; __syncthreads();
  for (int o = blockDim.x >> 1; o > 0; o >>= 1) { if (threadIdx.x < o) sh[threadIdx.x] = fmaxf(sh[threadIdx.x], sh[threadIdx.x + o]); __syncthreads(); }
  const float maxD = sh[0]; __syncthreads();
  float sum = 0.f;
  for (int j = threadIdx.x; j < S; j += blockDim.x)
    if (j <= i) sum += row[j] * expf(di - dc[j] + igr[j] - maxD);
  sh[threadIdx.x] = sum; __syncthreads();
  for (int o = blockDim.x >> 1; o > 0; o >>= 1) { if (threadIdx.x < o) sh[threadIdx.x] += sh[threadIdx.x + o]; __syncthreads(); }
  const float norm = fmaxf(fabsf(sh[0]), expf(-maxD)) + 1e-6f; __syncthreads();
  for (int j = threadIdx.x; j < S; j += blockDim.x) {
    float v = (j <= i) ? row[j] * expf(di - dc[j] + igr[j] - maxD) : 0.f;
    row[j] = v / norm;
  }
}

__global__ void k_combine(const float* __restrict__ hn, const float* __restrict__ xc,
                          const float* __restrict__ z, const float* __restrict__ skip,
                          float* __restrict__ out, long total, int C)
{
  GSTRIDE(i, total) {
    int c = (int)(i % C);
    out[i] = (hn[i] + skip[c] * xc[i]) * siluf(z[i]);
  }
}

__global__ void k_geglu(const float* __restrict__ u2, float* __restrict__ out, long rows, int pf)
{
  GSTRIDE(i, rows * pf) {
    long r = i / pf; int c = (int)(i % pf);
    float g = u2[r * 2 * pf + c];
    float u = u2[r * 2 * pf + pf + c];
    out[i] = geluf(g) * u;
  }
}

__global__ void k_gelu(const float* __restrict__ x, float* __restrict__ y, long n)
{ GSTRIDE(i, n) y[i] = geluf(x[i]); }

__global__ void k_add3(const float* __restrict__ a, const float* __restrict__ b,
                       float* __restrict__ o, long n)
{ GSTRIDE(i, n) o[i] = a[i] + b[i]; }

__global__ void k_addip(float* __restrict__ a, const float* __restrict__ b, long n)
{ GSTRIDE(i, n) a[i] += b[i]; }

__global__ void k_copy_h0(const float* __restrict__ enc, float* __restrict__ h)
{
  GSTRIDE(i, (long)NB32 * EMB) {
    int b = (int)(i / EMB); int c = (int)(i % EMB);
    h[i] = enc[((long)b * SWIN + (SWIN - 1)) * EMB + c];
  }
}

// ---------------- host-side plumbing ----------------
struct MlstmP {
  const float *ln_w, *proj_up_w, *conv_w, *conv_b, *q_w, *k_w, *v_w,
              *ig_w, *ig_b, *fg_w, *fg_b, *outnorm_w, *skip, *proj_down_w;
};
struct SlstmP {
  const float *ln1_w, *conv_w, *conv_b, *gin_w, *rec_w, *gate_b,
              *gn_w, *ln2_w, *ffn_up_w, *ffn_down_w;
};
struct StackP { MlstmP b0; SlstmP b1; MlstmP b2; const float* post_ln_w; };

struct WS {
  float *tln, *t640, *xe, *xwork, *hcur, *gbuf;
  float *xm, *z, *xc, *q, *k, *v, *qh, *kh, *vh;
  float *ig, *fg, *dcum, *Cmat, *hatt, *hn, *comb;
  float *xc2, *gx, *ys;
  float *u2, *gg;
};

static inline dim3 grid1d(long n) {
  long b = (n + 255) / 256; if (b > 65535) b = 65535; if (b < 1) b = 1;
  return dim3((unsigned)b);
}

static void run_gemm(hipStream_t st,
                     const float* A, long sA, int lda,
                     const float* B, long sB, int ldb, int bKN,
                     float* C, long sC, int ldc,
                     const float* bias, long sBias,
                     int M, int N, int K, int batch, float alpha)
{
  dim3 g((N + GT_N - 1) / GT_N, (M + GT_M - 1) / GT_M, batch);
  k_gemm<<<g, 128, 0, st>>>(A, sA, lda, B, sB, ldb, bKN, C, sC, ldc, bias, sBias, M, N, K, alpha);
}

static void run_ln(hipStream_t st, const float* x, const float* w, float* y, int rows)
{ k_layernorm<<<rows, 256, 0, st>>>(x, w, y, EMB); }

static void mlstm_forward(hipStream_t st, const float* xn, int rows, int S,
                          const MlstmP& p, WS& w, float* outb)
{
  const int B = rows / S;
  const long ne = (long)rows * DINNER;
  // proj_up split -> xm, z
  run_gemm(st, xn, 0, EMB, p.proj_up_w,                     0, EMB, 0, w.xm, 0, DINNER, nullptr, 0, rows, DINNER, EMB, 1, 1.f);
  run_gemm(st, xn, 0, EMB, p.proj_up_w + (long)DINNER*EMB,  0, EMB, 0, w.z,  0, DINNER, nullptr, 0, rows, DINNER, EMB, 1, 1.f);
  k_conv_silu<<<grid1d(ne), 256, 0, st>>>(w.xm, p.conv_w, p.conv_b, w.xc, B, S, DINNER, 8);
  k_headwise<<<grid1d(ne), 256, 0, st>>>(w.xc, p.q_w, w.q, rows, NBQ, BSQ, BSQ);
  k_headwise<<<grid1d(ne), 256, 0, st>>>(w.xc, p.k_w, w.k, rows, NBQ, BSQ, BSQ);
  k_headwise<<<grid1d(ne), 256, 0, st>>>(w.xm, p.v_w, w.v, rows, NBQ, BSQ, BSQ);
  k_mlstm_gates<<<rows, 256, 0, st>>>(w.q, w.k, w.v, p.ig_w, p.ig_b, p.fg_w, p.fg_b, w.ig, w.fg, S, DINNER);
  k_cumlogsig<<<(B * NHM + 127) / 128, 128, 0, st>>>(w.fg, w.dcum, B * NHM, S);
  k_pack_heads<<<grid1d(ne), 256, 0, st>>>(w.q, w.qh, B, S, NHM, DHM);
  k_pack_heads<<<grid1d(ne), 256, 0, st>>>(w.k, w.kh, B, S, NHM, DHM);
  k_pack_heads<<<grid1d(ne), 256, 0, st>>>(w.v, w.vh, B, S, NHM, DHM);
  // C = q k^T * DH^-0.5   (batched over B*NHM)
  run_gemm(st, w.qh, (long)S * DHM, DHM, w.kh, (long)S * DHM, DHM, 0,
           w.Cmat, (long)S * S, S, nullptr, 0, S, S, DHM, B * NHM, 1.f / sqrtf((float)DHM));
  k_attn_norm<<<B * NHM * S, 128, 0, st>>>(w.Cmat, w.dcum, w.ig, S);
  // h = Cn @ v   (B stored K x N -> bKN=1)
  run_gemm(st, w.Cmat, (long)S * S, S, w.vh, (long)S * DHM, DHM, 1,
           w.hatt, (long)S * DHM, DHM, nullptr, 0, S, DHM, S, B * NHM, 1.f);
  k_mhnorm<<<rows * NHM, 128, 0, st>>>(w.hatt, p.outnorm_w, w.hn, S, NHM, DHM, 0);
  k_combine<<<grid1d(ne), 256, 0, st>>>(w.hn, w.xc, w.z, p.skip, w.comb, ne, DINNER);
  run_gemm(st, w.comb, 0, DINNER, p.proj_down_w, 0, DINNER, 0, outb, 0, EMB, nullptr, 0, rows, EMB, DINNER, 1, 1.f);
}

static void slstm_forward(hipStream_t st, const float* xn, int rows, int S,
                          const SlstmP& p, WS& w, float* outb)
{
  const int B = rows / S;
  const long ne = (long)rows * EMB;
  k_conv_silu<<<grid1d(ne), 256, 0, st>>>(xn, p.conv_w, p.conv_b, w.xc2, B, S, EMB, 4);
  for (int g = 0; g < 4; ++g) {
    const float* A = (g < 2) ? w.xc2 : xn;   // gi,gf from conv; gz,go from x
    run_gemm(st, A, 160, EMB,
             p.gin_w + (long)g * NHS * DHS * DHS, (long)DHS * DHS, DHS, 0,
             w.gx + (long)g * NHS * rows * DHS, (long)rows * DHS, DHS,
             p.gate_b + (long)g * NHS * DHS, DHS,
             rows, DHS, DHS, NHS, 1.f);
  }
  size_t dyn = (size_t)(4 * DHS * DHS + NB32 * DHS) * sizeof(_Float16);
  k_slstm_scan<<<NHS, 320, dyn, st>>>(w.gx, p.rec_w, w.ys, S);
  k_mhnorm<<<rows * NHS, 128, 0, st>>>(w.ys, p.gn_w, outb, S, NHS, DHS, 1);
}

static void ffn_forward(hipStream_t st, const float* xn, int rows,
                        const SlstmP& p, WS& w, float* outb)
{
  run_gemm(st, xn, 0, EMB, p.ffn_up_w, 0, EMB, 0, w.u2, 0, 2 * PFF, nullptr, 0, rows, 2 * PFF, EMB, 1, 1.f);
  k_geglu<<<grid1d((long)rows * PFF), 256, 0, st>>>(w.u2, w.gg, rows, PFF);
  run_gemm(st, w.gg, 0, PFF, p.ffn_down_w, 0, PFF, 0, outb, 0, EMB, nullptr, 0, rows, EMB, PFF, 1, 1.f);
}

static void stack_fwd(hipStream_t st, const float* xin, float* xwork, float* xout,
                      int rows, int S, const StackP& sp, WS& w)
{
  const long ne = (long)rows * EMB;
  run_ln(st, xin, sp.b0.ln_w, w.tln, rows);
  mlstm_forward(st, w.tln, rows, S, sp.b0, w, w.t640);
  k_add3<<<grid1d(ne), 256, 0, st>>>(xin, w.t640, xwork, ne);

  run_ln(st, xwork, sp.b1.ln1_w, w.tln, rows);
  slstm_forward(st, w.tln, rows, S, sp.b1, w, w.t640);
  k_addip<<<grid1d(ne), 256, 0, st>>>(xwork, w.t640, ne);

  run_ln(st, xwork, sp.b1.ln2_w, w.tln, rows);
  ffn_forward(st, w.tln, rows, sp.b1, w, w.t640);
  k_addip<<<grid1d(ne), 256, 0, st>>>(xwork, w.t640, ne);

  run_ln(st, xwork, sp.b2.ln_w, w.tln, rows);
  mlstm_forward(st, w.tln, rows, S, sp.b2, w, w.t640);
  k_addip<<<grid1d(ne), 256, 0, st>>>(xwork, w.t640, ne);

  run_ln(st, xwork, sp.post_ln_w, xout, rows);
}

static MlstmP parseM(void* const* d, int& i) {
  MlstmP p;
  p.ln_w        = (const float*)d[i++]; p.proj_up_w   = (const float*)d[i++];
  p.conv_w      = (const float*)d[i++]; p.conv_b      = (const float*)d[i++];
  p.q_w         = (const float*)d[i++]; p.k_w         = (const float*)d[i++];
  p.v_w         = (const float*)d[i++]; p.ig_w        = (const float*)d[i++];
  p.ig_b        = (const float*)d[i++]; p.fg_w        = (const float*)d[i++];
  p.fg_b        = (const float*)d[i++]; p.outnorm_w   = (const float*)d[i++];
  p.skip        = (const float*)d[i++]; p.proj_down_w = (const float*)d[i++];
  return p;
}
static SlstmP parseS(void* const* d, int& i) {
  SlstmP p;
  p.ln1_w     = (const float*)d[i++]; p.conv_w     = (const float*)d[i++];
  p.conv_b    = (const float*)d[i++]; p.gin_w      = (const float*)d[i++];
  p.rec_w     = (const float*)d[i++]; p.gate_b     = (const float*)d[i++];
  p.gn_w      = (const float*)d[i++]; p.ln2_w      = (const float*)d[i++];
  p.ffn_up_w  = (const float*)d[i++]; p.ffn_down_w = (const float*)d[i++];
  return p;
}
static StackP parseStack(void* const* d, int& i) {
  StackP s;
  s.b0 = parseM(d, i); s.b1 = parseS(d, i); s.b2 = parseM(d, i);
  s.post_ln_w = (const float*)d[i++];
  return s;
}

extern "C" void kernel_launch(void* const* d_in, const int* in_sizes, int n_in,
                              void* d_out, int out_size, void* d_ws, size_t ws_size,
                              hipStream_t stream)
{
  (void)in_sizes; (void)n_in; (void)out_size; (void)ws_size;
  // setup_inputs() dict order: src, then params in construction order.
  int i = 0;
  const float* src   = (const float*)d_in[i++];
  const float* enc_w = (const float*)d_in[i++];
  const float* enc_b = (const float*)d_in[i++];
  StackP enc = parseStack(d_in, i);
  StackP dec = parseStack(d_in, i);
  const float* fc_w = (const float*)d_in[i++];
  const float* fc_b = (const float*)d_in[i++];

  // workspace arena
  char* cur = (char*)d_ws;
  auto carve = [&](size_t nfloats) -> float* {
    float* r = (float*)cur;
    cur += ((nfloats * sizeof(float) + 255) & ~(size_t)255);
    return r;
  };
  const long RE = (long)NB32 * SWIN;   // 4096 encoder rows
  WS w;
  w.tln   = carve((size_t)RE * EMB);
  w.t640  = carve((size_t)RE * EMB);
  w.xe    = carve((size_t)RE * EMB);
  w.xwork = carve((size_t)RE * EMB);
  w.hcur  = carve((size_t)NB32 * EMB);
  w.gbuf  = carve((size_t)NB32 * EMB);
  w.xm    = carve((size_t)RE * DINNER);
  w.z     = carve((size_t)RE * DINNER);
  w.xc    = carve((size_t)RE * DINNER);
  w.q     = carve((size_t)RE * DINNER);
  w.k     = carve((size_t)RE * DINNER);
  w.v     = carve((size_t)RE * DINNER);
  w.qh    = carve((size_t)RE * DINNER);
  w.kh    = carve((size_t)RE * DINNER);
  w.vh    = carve((size_t)RE * DINNER);
  w.ig    = carve((size_t)NB32 * NHM * SWIN);
  w.fg    = carve((size_t)NB32 * NHM * SWIN);
  w.dcum  = carve((size_t)NB32 * NHM * SWIN);
  w.Cmat  = carve((size_t)NB32 * NHM * SWIN * SWIN);
  w.hatt  = carve((size_t)NB32 * NHM * SWIN * DHM);
  w.hn    = carve((size_t)RE * DINNER);
  w.comb  = carve((size_t)RE * DINNER);
  w.xc2   = carve((size_t)RE * EMB);
  w.gx    = carve((size_t)4 * NHS * RE * DHS);
  w.ys    = carve((size_t)RE * EMB);
  w.u2    = carve((size_t)RE * 2 * PFF);
  w.gg    = carve((size_t)RE * PFF);

  (void)hipFuncSetAttribute((const void*)k_slstm_scan,
                            hipFuncAttributeMaxDynamicSharedMemorySize,
                            (int)((4 * DHS * DHS + NB32 * DHS) * sizeof(_Float16)));

  // ---------- encoder ----------
  run_gemm(stream, src, 0, NFEATS, enc_w, 0, NFEATS, 0, w.xe, 0, EMB,
           enc_b, 0, (int)RE, EMB, NFEATS, 1, 1.f);
  stack_fwd(stream, w.xe, w.xwork, w.xe, (int)RE, SWIN, enc, w);
  k_copy_h0<<<grid1d((long)NB32 * EMB), 256, 0, stream>>>(w.xe, w.hcur);

  // ---------- autoregressive decoder (128 sequential steps, S=1) ----------
  float* out = (float*)d_out;
  for (int t = 0; t < SWIN; ++t) {
    stack_fwd(stream, w.hcur, w.xwork, w.hcur, NB32, 1, dec, w);
    k_gelu<<<grid1d((long)NB32 * EMB), 256, 0, stream>>>(w.hcur, w.gbuf, (long)NB32 * EMB);
    run_gemm(stream, w.gbuf, 0, EMB, fc_w, 0, EMB, 0,
             out + (long)t * NB32 * NFEATS, 0, NFEATS,
             fc_b, 0, NB32, NFEATS, EMB, 1, 1.f);
  }
}
